// SpaceTimeLookTable_57939108823486
// MI455X (gfx1250) — matrix operand order repaired and checked by reading, not compiled, not run on hardware
//
#include <hip/hip_runtime.h>
#include <hip/hip_bf16.h>

typedef __bf16 bf16_t;
typedef __bf16 v8bf  __attribute__((ext_vector_type(8)));
typedef __bf16 v16bf __attribute__((ext_vector_type(16)));
typedef float  v8f   __attribute__((ext_vector_type(8)));

#define NSAMP   8192
#define KTOT    12240
#define KPAD    12288
#define KPANEL  128
#define NPANEL  96          // KPAD / KPANEL
#define INNERD  64
#define ROWS    16
#define LROW    136         // 128 bf16 + 8 pad  (272 B stride, 16B aligned)
#define NPAIR   172
#define DESC_INVALID 0xFFFFFFFFu
#define WS_DESC_BYTES 49152 // 12288 * 4, 16B aligned

// 6 face offsets then 8 corner offsets (reference order)
__constant__ int SPAT14[14][3] = {
  { 1,0,0},{-1,0,0},{0, 1,0},{0,-1,0},{0,0, 1},{0,0,-1},
  { 1, 1, 1},{-1,-1,-1},{ 1,-1,-1},{-1, 1, 1},
  { 1, 1,-1},{-1,-1, 1},{ 1,-1, 1},{-1, 1,-1}};
__constant__ int EXPT[11] = {-64,-8,-4,-2,-1,0,1,2,4,8,64};

// ---------------- prologue 1: feature-index -> (table, pair, channel) ----------
__global__ void k_init_desc(unsigned* __restrict__ desc) {
  int k = blockIdx.x * 256 + threadIdx.x;
  if (k >= KPAD) return;
  unsigned d = DESC_INVALID;
  if      (k < 112)   { int r = k;        d = (unsigned)(0  + r/16)   | ((unsigned)(r%16)  << 8) | (0u << 20); }
  else if (k < 560)   { int r = k-112;    d = (unsigned)(7  + r/64)   | ((unsigned)(r%64)  << 8) | (1u << 20); }
  else if (k < 2352)  { int r = k-560;    d = (unsigned)(14 + r/256)  | ((unsigned)(r%256) << 8) | (2u << 20); }
  else if (k < 9520)  { int r = k-2352;   d = (unsigned)(21 + r/1024) | ((unsigned)(r%1024)<< 8) | (3u << 20); }
  else if (k < 11312) { int r = k-9520;   d = (unsigned)(28 + r/64)   | ((unsigned)(r%64)  << 8) | (4u << 20); }
  else if (k < 11536) { int r = k-11312;  d = (unsigned)(56 + r/8)    | ((unsigned)(r%8)   << 8) | (5u << 20); }
  else if (k < KTOT)  { int r = k-11536;  d = (unsigned)(84 + r/8)    | ((unsigned)(r%8)   << 8) | (6u << 20); }
  desc[k] = d;
}

// ---------------- prologue 2: wT[n][k] = bf16(rms_scale[k] * w_down[k][n]) -----
__global__ void k_init_wt(const float* __restrict__ w_down,
                          const float* __restrict__ rms_scale,
                          bf16_t* __restrict__ wT) {
  long g = (long)blockIdx.x * 256 + threadIdx.x;
  if (g >= (long)INNERD * KPAD) return;
  int n = (int)(g / KPAD);
  int k = (int)(g % KPAD);
  float v = 0.f;
  if (k < KTOT) v = rms_scale[k] * w_down[(long)k * INNERD + n];
  wT[(long)n * KPAD + k] = (bf16_t)v;
}

// ---------------- fused gather + WMMA GEMM + head ------------------------------
__global__ __launch_bounds__(128) void k_main(
    const float* __restrict__ pos, const float* __restrict__ dirv, const float* __restrict__ tval,
    const float* __restrict__ tab0, const float* __restrict__ tab1,
    const float* __restrict__ tab2, const float* __restrict__ tab3,
    const float* __restrict__ st1,  const float* __restrict__ st2,  const float* __restrict__ tf3,
    const float* __restrict__ w_final, const float* __restrict__ b_final,
    const unsigned* __restrict__ desc, const bf16_t* __restrict__ wT,
    float* __restrict__ out)
{
  __shared__ float  posL[ROWS][3];
  __shared__ float  dirL[ROWS][3];
  __shared__ float  tL[ROWS];
  __shared__ int    baseOff[ROWS][NPAIR];
  __shared__ const float* tabs[8];
  __shared__ bf16_t Alds[ROWS * LROW];     // A panel: 16 rows x 128 K (bf16)
  __shared__ bf16_t Blds[64 * LROW];       // B panel: 64 cols x 128 K (bf16, col-major)
  __shared__ float  Clds[ROWS][68];        // result tile 16 x 64 (padded)
  __shared__ float  ssqArr[128];
  __shared__ float  invL[ROWS];
  __shared__ float  wfL[68 * 4 + 4];       // w_final (68x4) then b_final (4)

  const int tid  = threadIdx.x;
  const int tile = blockIdx.x;
  const int base = tile * ROWS;

  if (tid < ROWS)                 tL[tid] = tval[base + tid];
  if (tid >= 32 && tid < 32 + ROWS*3) { int i = tid - 32; posL[i/3][i%3] = pos [base*3 + i]; }
  if (tid >= 80 && tid < 80 + ROWS*3) { int i = tid - 80; dirL[i/3][i%3] = dirv[base*3 + i]; }
  if (tid == 0) {
    tabs[0]=tab0; tabs[1]=tab1; tabs[2]=tab2; tabs[3]=tab3;
    tabs[4]=st1;  tabs[5]=st2;  tabs[6]=tf3;  tabs[7]=tab0;
  }
  for (int i = tid; i < 68*4; i += 128) wfL[i] = w_final[i];
  if (tid < 4) wfL[68*4 + tid] = b_final[tid];
  __syncthreads();

  // ---- per-sample neighbor base offsets (element offsets incl. feature stride)
  for (int i = tid; i < ROWS * NPAIR; i += 128) {
    int m = i / NPAIR, p = i % NPAIR;
    int off;
    if (p < 28) {                                 // spatial tables 0..3: center + 6 faces
      int tb = p / 7, nb = p % 7;
      int D = 128 >> tb, Dm = D - 1;
      int F = 16 << (2 * tb);
      int ix = (int)(posL[m][0] * (float)Dm);
      int iy = (int)(posL[m][1] * (float)Dm);
      int iz = (int)(posL[m][2] * (float)Dm);
      int ox = 0, oy = 0, oz = 0;
      if (nb) { ox = SPAT14[nb-1][0]; oy = SPAT14[nb-1][1]; oz = SPAT14[nb-1][2]; }
      int x = (ix + ox) & Dm, y = (iy + oy) & Dm, z = (iz + oz) & Dm;
      off = ((x * D + y) * D + z) * F;
    } else {
      float tv = tL[m];
      // reference quirk: t_idx built from DIMS_ST1 is reused for st2 and tf3
      int i0 = (int)(tv * 15.f);
      int i3 = (int)(tv * 63.f);
      if (p < 56) {                               // st1: dims (16,16,16,64), F=64
        int r = p - 28;
        const int* s = SPAT14[r >> 1];
        int ta = (r & 1) ? -1 : 1;
        int n0=(i0+s[0])&15, n1=(i0+s[1])&15, n2=(i0+s[2])&15, n3=(i3+ta)&63;
        off = (((n0*16 + n1)*16 + n2)*64 + n3) * 64;
      } else if (p < 84) {                        // st2: dims (64,)*4, F=8
        int r = p - 56;
        const int* s = SPAT14[r >> 1];
        int ta = (r & 1) ? -1 : 1;
        int n0=(i0+s[0])&63, n1=(i0+s[1])&63, n2=(i0+s[2])&63, n3=(i3+ta)&63;
        off = (((n0*64 + n1)*64 + n2)*64 + n3) * 8;
      } else {                                    // tf3: dims (4,4,4,65536), F=8
        int r = p - 84;
        const int* s = SPAT14[6 + r/11];
        int ta = EXPT[r % 11];
        int n0=(i0+s[0])&3, n1=(i0+s[1])&3, n2=(i0+s[2])&3, n3=(i3+ta)&65535;
        off = (((n0*4 + n1)*4 + n2)*65536 + n3) * 8;
      }
    }
    baseOff[m][p] = off;
  }
  __syncthreads();

  const int row  = tid >> 3;          // sample row handled by this thread (gather)
  const int l8   = tid & 7;
  const int lane = tid & 31;
  const int wv   = tid >> 5;          // wave id = N-tile id (0..3)
  const int mA   = lane & 15;
  const int hoA  = (lane < 16) ? 0 : 8;    // A: lanes 0-15 K[0:8)+[16:24); 16-31 K[8:16)+[24:32)
  const int hoB  = (lane < 16) ? 0 : 8;    // B: lanes 0-15 K[0:8)+[16:24); 16-31 K[8:16)+[24:32)

  float ssq = 0.f;
  v8f c = {0.f,0.f,0.f,0.f,0.f,0.f,0.f,0.f};

  for (int pnl = 0; pnl < NPANEL; ++pnl) {
    const int k0 = pnl * KPANEL;
    // ---- stream B panel from bf16 w_downT (L2-resident) into LDS, col-major
    #pragma unroll
    for (int it = 0; it < 8; ++it) {
      int ci = tid + it * 128;
      int col = ci >> 4, cp = ci & 15;
      const uint4 v = *(const uint4*)(wT + (long)col * KPAD + k0 + cp * 8);
      *(uint4*)(&Blds[col * LROW + cp * 8]) = v;
    }
    if (pnl + 1 < NPANEL)   // warm next weight panel (global_prefetch_b8)
      __builtin_prefetch(wT + (long)(tid >> 1) * KPAD + k0 + KPANEL + (tid & 1) * 64, 0, 1);

    // ---- gather A panel: relu(table lookup), accumulate ssq, store bf16
    #pragma unroll
    for (int j = 0; j < 16; ++j) {
      int k = l8 + j * 8;
      unsigned d = desc[k0 + k];
      float v = 0.f;
      if (d != DESC_INVALID) {
        int pair = d & 255;
        int ch   = (d >> 8) & 1023;
        int tb   = d >> 20;
        v = tabs[tb][baseOff[row][pair] + ch];
        v = fmaxf(v, 0.f);
      }
      ssq += v * v;
      Alds[row * LROW + k] = (bf16_t)v;
    }
    __syncthreads();

    // ---- 4 x K=32 WMMA steps; each wave owns one 16-wide N tile
    #pragma unroll
    for (int kk = 0; kk < 4; ++kk) {
      const bf16_t* ap = &Alds[mA * LROW + kk * 32 + hoA];
      v8bf a0 = *(const v8bf*)ap;
      v8bf a1 = *(const v8bf*)(ap + 16);
      const bf16_t* bp = &Blds[(mA + 16 * wv) * LROW + kk * 32 + hoB];
      v8bf b0 = *(const v8bf*)bp;
      v8bf b1 = *(const v8bf*)(bp + 16);
      v16bf a, b;
      #pragma unroll
      for (int e = 0; e < 8; ++e) { a[e]=a0[e]; a[e+8]=a1[e]; b[e]=b0[e]; b[e+8]=b1[e]; }
      c = __builtin_amdgcn_wmma_f32_16x16x32_bf16(false, a, false, b, (short)0, c, false, false);
    }
    __syncthreads();
  }

  // ---- epilogue: ssq reduce -> 1/(rms+eps); C -> LDS; 68x4 head + sigmoid
  ssqArr[tid] = ssq;
  #pragma unroll
  for (int v = 0; v < 8; ++v) {
    int m = v + ((lane < 16) ? 0 : 8);       // C layout: VGPR v, lanes<16 -> M=v else M=v+8
    Clds[m][wv * 16 + mA] = c[v];
  }
  __syncthreads();
  if (tid < ROWS) {
    float s = 0.f;
    #pragma unroll
    for (int q = 0; q < 8; ++q) s += ssqArr[tid * 8 + q];
    float rms = sqrtf(s) * 0.00903877f;      // * 12240^-0.5
    invL[tid] = 1.f / (rms + 1e-8f);
  }
  __syncthreads();
  if (tid < 64) {
    int m = tid >> 2, o = tid & 3;
    float inv = invL[m];
    float acc = wfL[68 * 4 + o];
    #pragma unroll 8
    for (int j = 0; j < 64; ++j) {
      float h = fmaxf(Clds[m][j] * inv, 0.f);
      acc += h * wfL[j * 4 + o];
    }
    acc += dirL[m][0] * wfL[64*4 + o] + dirL[m][1] * wfL[65*4 + o]
         + dirL[m][2] * wfL[66*4 + o] + tL[m]      * wfL[67*4 + o];
    out[(long)(base + m) * 4 + o] = 1.f / (1.f + __expf(-acc));
  }
}

extern "C" void kernel_launch(void* const* d_in, const int* in_sizes, int n_in,
                              void* d_out, int out_size, void* d_ws, size_t ws_size,
                              hipStream_t stream) {
  const float* pos  = (const float*)d_in[0];
  const float* dir  = (const float*)d_in[1];
  const float* t    = (const float*)d_in[2];
  const float* tab0 = (const float*)d_in[3];
  const float* tab1 = (const float*)d_in[4];
  const float* tab2 = (const float*)d_in[5];
  const float* tab3 = (const float*)d_in[6];
  const float* st1  = (const float*)d_in[7];
  const float* st2  = (const float*)d_in[8];
  const float* tf3  = (const float*)d_in[9];
  const float* rms  = (const float*)d_in[10];
  const float* wdn  = (const float*)d_in[11];
  const float* wfn  = (const float*)d_in[12];
  const float* bfn  = (const float*)d_in[13];
  (void)in_sizes; (void)n_in; (void)out_size; (void)ws_size;

  unsigned* desc = (unsigned*)d_ws;
  bf16_t*   wT   = (bf16_t*)((char*)d_ws + WS_DESC_BYTES);

  k_init_desc<<<KPAD / 256, 256, 0, stream>>>(desc);
  k_init_wt<<<(INNERD * KPAD) / 256, 256, 0, stream>>>(wdn, rms, wT);
  k_main<<<NSAMP / ROWS, 128, 0, stream>>>(pos, dir, t, tab0, tab1, tab2, tab3,
                                           st1, st2, tf3, wfn, bfn, desc, wT,
                                           (float*)d_out);
}
